// AutoSeg_34608846471703
// MI455X (gfx1250) — matrix operand order repaired and checked by reading, not compiled, not run on hardware
//
#include <hip/hip_runtime.h>
#include <math.h>

typedef __bf16 bf16_t;
typedef bf16_t v16bf __attribute__((ext_vector_type(16)));
typedef bf16_t v8bf  __attribute__((ext_vector_type(8)));
typedef float  v8f   __attribute__((ext_vector_type(8)));
typedef float  v4f   __attribute__((ext_vector_type(4)));

#define NEGBIG (-1e9f)

// ============================================================
// Batched GEMM with bf16 WMMA:  C = act(scale * A @ B(^T) + bias + res)
//   A: [batch][M][K] f32
//   B: bTrans==0 -> [batch][N][K]  (X @ W^T, torch Linear weight layout)
//      bTrans==1 -> [batch][K][N]
//   bias: [N] or null ; res: [batch][M][N] (stride sC) or null
//   act: 0=none 1=quick_gelu 2=erf gelu
// Block = 256 threads = 8 waves; 64x64 tile; each wave 16x32 (2 WMMAs).
// ============================================================
__launch_bounds__(256)
__global__ void k_gemm(const float* __restrict__ A, const float* __restrict__ Bm,
                       const float* __restrict__ bias, const float* __restrict__ res,
                       float* __restrict__ C,
                       int M, int N, int K,
                       long long sA, long long sB, long long sC,
                       int bTrans, float scale, int act)
{
    __shared__ bf16_t As[64][40];
    __shared__ bf16_t Bs[64][40];

    const int batch = blockIdx.z;
    const float* Ab = A  + (long long)batch * sA;
    const float* Bb = Bm + (long long)batch * sB;
    float*       Cb = C  + (long long)batch * sC;
    const float* Rb = res ? res + (long long)batch * sC : nullptr;

    const int tid  = threadIdx.x;
    const int lane = tid & 31;
    const int wave = tid >> 5;      // 0..7
    const int wm   = wave >> 1;     // 0..3 -> M offset 16*wm
    const int wn   = wave & 1;      // 0..1 -> N offset 32*wn
    const int bm   = blockIdx.y * 64;
    const int bn   = blockIdx.x * 64;

    const int row = tid >> 2;       // 0..63
    const int kg  = (tid & 3) * 8;  // 0,8,16,24

    v8f acc0 = {}, acc1 = {};

    const bool kVec  = ((K & 3) == 0);
    const bool aFull = (bm + 64 <= M) && kVec;
    const bool bFull = (bn + 64 <= N) && kVec && (bTrans == 0);

    for (int k0 = 0; k0 < K; k0 += 32) {
        __syncthreads();
        const bool kFull = (k0 + 32 <= K);

        // ---------- stage A tile [64 x 32] as bf16 ----------
        {
            const int gm = bm + row;
            v8bf h;
            if (aFull && kFull) {
                const v4f* p4 = (const v4f*)&Ab[(long long)gm * K + k0 + kg];
                __builtin_prefetch(&Ab[(long long)gm * K + k0 + 32 + kg], 0, 0);
                v4f f0 = p4[0], f1 = p4[1];
                h[0]=(bf16_t)f0.x; h[1]=(bf16_t)f0.y; h[2]=(bf16_t)f0.z; h[3]=(bf16_t)f0.w;
                h[4]=(bf16_t)f1.x; h[5]=(bf16_t)f1.y; h[6]=(bf16_t)f1.z; h[7]=(bf16_t)f1.w;
            } else {
                const int gmc = gm < M ? gm : M - 1;
                #pragma unroll
                for (int j = 0; j < 8; ++j) {
                    const int gk  = k0 + kg + j;
                    const int gkc = gk < K ? gk : K - 1;
                    const float v = Ab[(long long)gmc * K + gkc];
                    h[j] = (bf16_t)((gm < M && gk < K) ? v : 0.0f);
                }
            }
            *(v8bf*)&As[row][kg] = h;
        }
        // ---------- stage B tile: Bs[n][k] ----------
        {
            const int gn = bn + row;
            v8bf h;
            if (bFull && kFull) {
                const v4f* p4 = (const v4f*)&Bb[(long long)gn * K + k0 + kg];
                v4f f0 = p4[0], f1 = p4[1];
                h[0]=(bf16_t)f0.x; h[1]=(bf16_t)f0.y; h[2]=(bf16_t)f0.z; h[3]=(bf16_t)f0.w;
                h[4]=(bf16_t)f1.x; h[5]=(bf16_t)f1.y; h[6]=(bf16_t)f1.z; h[7]=(bf16_t)f1.w;
            } else {
                const int gnc = gn < N ? gn : N - 1;
                #pragma unroll
                for (int j = 0; j < 8; ++j) {
                    const int gk  = k0 + kg + j;
                    const int gkc = gk < K ? gk : K - 1;
                    const long long idx = bTrans ? (long long)gkc * N + gnc
                                                 : (long long)gnc * K + gkc;
                    const float v = Bb[idx];
                    h[j] = (bf16_t)((gn < N && gk < K) ? v : 0.0f);
                }
            }
            *(v8bf*)&Bs[row][kg] = h;
        }
        __syncthreads();

        // ---------- fragments per ISA VGPR layout ----------
        const int half = lane >> 4;
        const int mr   = lane & 15;
        v16bf af, bf0, bf1;
        const bf16_t* ap  = &As[wm * 16 + mr][half * 8];
        #pragma unroll
        for (int e = 0; e < 8; ++e) { af[e] = ap[e]; af[e + 8] = ap[e + 16]; }
        const bf16_t* bp0 = &Bs[wn * 32 + mr][half * 8];
        const bf16_t* bp1 = &Bs[wn * 32 + 16 + mr][half * 8];
        #pragma unroll
        for (int e = 0; e < 8; ++e) { bf0[e] = bp0[e]; bf0[e + 8] = bp0[e + 16]; bf1[e] = bp1[e]; bf1[e + 8] = bp1[e + 16]; }

        acc0 = __builtin_amdgcn_wmma_f32_16x16x32_bf16(false, af, false, bf0, (short)0, acc0, false, false);
        acc1 = __builtin_amdgcn_wmma_f32_16x16x32_bf16(false, af, false, bf1, (short)0, acc1, false, false);
    }

    // ---- epilogue: VGPR r -> M = r + 8*half, N = lane&15 ----
    const int half = lane >> 4;
    const int n0 = bn + wn * 32 + (lane & 15);
    const int n1 = n0 + 16;
    #pragma unroll
    for (int r = 0; r < 8; ++r) {
        const int m = bm + wm * 16 + r + half * 8;
        if (m >= M) continue;
        if (n0 < N) {
            float v = acc0[r] * scale;
            if (bias) v += bias[n0];
            if (Rb)   v += Rb[(long long)m * N + n0];
            if (act == 1)      v = v / (1.0f + __expf(-1.702f * v));
            else if (act == 2) v = 0.5f * v * (1.0f + erff(v * 0.70710678f));
            Cb[(long long)m * N + n0] = v;
        }
        if (n1 < N) {
            float v = acc1[r] * scale;
            if (bias) v += bias[n1];
            if (Rb)   v += Rb[(long long)m * N + n1];
            if (act == 1)      v = v / (1.0f + __expf(-1.702f * v));
            else if (act == 2) v = 0.5f * v * (1.0f + erff(v * 0.70710678f));
            Cb[(long long)m * N + n1] = v;
        }
    }
}

// ============================================================
// LayerNorm: one wave32 per row.
// ============================================================
__global__ void k_layernorm(const float* __restrict__ x, const float* __restrict__ g,
                            const float* __restrict__ b, float* __restrict__ y, int D)
{
    const long long rowid = blockIdx.x;
    const float* xr = x + rowid * D;
    float* yr = y + rowid * D;
    const int lane = threadIdx.x;
    float s = 0.f, s2 = 0.f;
    for (int i = lane; i < D; i += 32) { float v = xr[i]; s += v; s2 += v * v; }
    #pragma unroll
    for (int o = 16; o; o >>= 1) { s += __shfl_xor(s, o, 32); s2 += __shfl_xor(s2, o, 32); }
    const float m   = s / (float)D;
    const float var = s2 / (float)D - m * m;
    const float inv = rsqrtf(var + 1e-5f);
    for (int i = lane; i < D; i += 32)
        yr[i] = (xr[i] - m) * inv * g[i] + b[i];
}

// ============================================================
// Row softmax with optional causal mask; one wave per row.
// ============================================================
__global__ void k_softmax(float* __restrict__ sc, int S, int L, int causal)
{
    const long long row = blockIdx.x;
    const int q = (int)(row % S);
    float* xr = sc + row * (long long)L;
    const int lane = threadIdx.x;
    float mx = -3.4e38f;
    for (int j = lane; j < L; j += 32) {
        float v = xr[j];
        if (causal && j > q) v += NEGBIG;
        xr[j] = v;
        mx = fmaxf(mx, v);
    }
    #pragma unroll
    for (int o = 16; o; o >>= 1) mx = fmaxf(mx, __shfl_xor(mx, o, 32));
    float sum = 0.f;
    for (int j = lane; j < L; j += 32) { float e = __expf(xr[j] - mx); xr[j] = e; sum += e; }
    #pragma unroll
    for (int o = 16; o; o >>= 1) sum += __shfl_xor(sum, o, 32);
    const float inv = 1.0f / sum;
    for (int j = lane; j < L; j += 32) xr[j] *= inv;
}

// ============================================================
// Elementwise kernels
// ============================================================
__global__ void k_zero(float* a, long long n)
{ long long i = (long long)blockIdx.x * 256 + threadIdx.x; if (i < n) a[i] = 0.f; }

// Copy routed through LDS via CDNA5 async-to-LDS path (ASYNCcnt).
typedef __attribute__((address_space(3))) float lds_f;
__global__ void k_copy(const float* __restrict__ a, float* __restrict__ b, long long n)
{
    __shared__ float buf[256];
    long long i = (long long)blockIdx.x * 256 + threadIdx.x;
    if (i < n) {
        lds_f* lp = (lds_f*)&buf[threadIdx.x];
        const float* gp = &a[i];
        asm volatile("global_load_async_to_lds_b32 %0, %1, off"
                     :: "v"(lp), "v"(gp) : "memory");
    }
    asm volatile("s_wait_asynccnt 0x0" ::: "memory");
    __syncthreads();
    if (i < n) b[i] = buf[threadIdx.x];
}

// y[b,h,s,d] = x[b,s,h*hd+d]
__global__ void k_to_heads(const float* __restrict__ x, float* __restrict__ y,
                           int Bb, int S, int nh, int hd)
{
    long long i = (long long)blockIdx.x * 256 + threadIdx.x;
    long long tot = (long long)Bb * S * nh * hd;
    if (i >= tot) return;
    int d = (int)(i % hd); long long t = i / hd;
    int s = (int)(t % S);  t /= S;
    int h = (int)(t % nh); int b = (int)(t / nh);
    y[i] = x[((long long)(b * S + s) * nh + h) * hd + d];
}

// y[b,s,h*hd+d] = x[b,h,s,d]
__global__ void k_from_heads(const float* __restrict__ x, float* __restrict__ y,
                             int Bb, int S, int nh, int hd)
{
    long long i = (long long)blockIdx.x * 256 + threadIdx.x;
    long long tot = (long long)Bb * S * nh * hd;
    if (i >= tot) return;
    int D = nh * hd;
    int c = (int)(i % D); long long t = i / D;
    int s = (int)(t % S); int b = (int)(t / S);
    int h = c / hd, d = c % hd;
    y[i] = x[((long long)(b * nh + h) * S + s) * hd + d];
}

// patches [8,49,3072] from pixel [8,3,224,224]
__global__ void k_patchify(const float* __restrict__ px, float* __restrict__ out)
{
    long long i = (long long)blockIdx.x * 256 + threadIdx.x;
    long long tot = 8LL * 49 * 3072;
    if (i >= tot) return;
    int d = (int)(i % 3072); long long t = i / 3072;
    int p = (int)(t % 49); int b = (int)(t / 49);
    int c = d / 1024, r = (d % 1024) / 32, cc = d % 32;
    int g = p / 7, g2 = p % 7;
    out[i] = px[(((long long)b * 3 + c) * 224 + (g * 32 + r)) * 224 + (g2 * 32 + cc)];
}

// x[8,50,768] = concat(cls, patch_emb) + pos
__global__ void k_buildx(const float* __restrict__ pe, const float* __restrict__ cls,
                         const float* __restrict__ pos, float* __restrict__ x)
{
    long long i = (long long)blockIdx.x * 256 + threadIdx.x;
    long long tot = 8LL * 50 * 768;
    if (i >= tot) return;
    int d = (int)(i % 768); long long t = i / 768;
    int s = (int)(t % 50); int b = (int)(t / 50);
    float v = (s == 0) ? cls[d] : pe[((long long)b * 49 + (s - 1)) * 768 + d];
    x[i] = v + pos[s * 768 + d];
}

// x[8,16,512] = tok_emb[ids] + pos_emb[:16]
__global__ void k_embed(const int* __restrict__ ids, const float* __restrict__ tok,
                        const float* __restrict__ pos, float* __restrict__ x)
{
    long long i = (long long)blockIdx.x * 256 + threadIdx.x;
    long long tot = 8LL * 16 * 512;
    if (i >= tot) return;
    int d = (int)(i % 512); long long t = i / 512;
    int tt = (int)(t % 16); int b = (int)(t / 16);
    x[i] = tok[(long long)ids[b * 16 + tt] * 512 + d] + pos[tt * 512 + d];
}

// a[t,s,b,c] = (a + abase[(b*50+s),c]) * mul[(b*16+t),c] + add[(b*16+t),c]
__global__ void k_film(float* __restrict__ a, const float* __restrict__ abase,
                       const float* __restrict__ mul, const float* __restrict__ add)
{
    long long i = (long long)blockIdx.x * 256 + threadIdx.x;
    long long tot = 16LL * 50 * 8 * 64;
    if (i >= tot) return;
    int c = (int)(i & 63); long long t2 = i >> 6;
    int b = (int)(t2 & 7); t2 >>= 3;
    int s = (int)(t2 % 50); int t = (int)(t2 / 50);
    float v = a[i] + abase[(long long)(b * 50 + s) * 64 + c];
    long long mi = (long long)(b * 16 + t) * 64 + c;
    a[i] = v * mul[mi] + add[mi];
}

// im2col for 3x3 SAME conv over a[16,50,8,64] (drop s=0) -> col[128*49, 576]
__global__ void k_im2col(const float* __restrict__ a3, float* __restrict__ col)
{
    long long i = (long long)blockIdx.x * 256 + threadIdx.x;
    long long tot = 128LL * 49 * 576;
    if (i >= tot) return;
    int kk = (int)(i % 576); long long t2 = i / 576;
    int p = (int)(t2 % 49); int ib = (int)(t2 / 49);
    int c = kk / 9, r9 = kk % 9, ky = r9 / 3, kx = r9 % 3;
    int y = p / 7, x = p % 7;
    int t = ib / 8, b = ib % 8;
    int yy = y + ky - 1, xx = x + kx - 1;
    float v = 0.f;
    if (yy >= 0 && yy < 7 && xx >= 0 && xx < 7) {
        int s = 1 + yy * 7 + xx;
        v = a3[(((long long)t * 50 + s) * 8 + b) * 64 + c];
    }
    col[i] = v;
}

// pixel-shuffle scatter for conv_t k=4 (+bias +gelu), channels-last out [128,784,32]
__global__ void k_scatter_t1(const float* __restrict__ tmp1, const float* __restrict__ bias,
                             float* __restrict__ y2)
{
    long long i = (long long)blockIdx.x * 256 + threadIdx.x;
    long long tot = 128LL * 784 * 32;
    if (i >= tot) return;
    int o = (int)(i % 32); long long t2 = i / 32;
    int q = (int)(t2 % 784); int ib = (int)(t2 / 784);
    int hy = q / 28, wx = q % 28;
    int h = hy / 4, d = hy % 4, w = wx / 4, e = wx % 4;
    float v = tmp1[((long long)ib * 49 + h * 7 + w) * 512 + o * 16 + d * 4 + e] + bias[o];
    v = 0.5f * v * (1.0f + erff(v * 0.70710678f));
    y2[i] = v;
}

// pixel-shuffle scatter for conv_t k=8 (+bias) into masks output [B,T,224,224]
__global__ void k_scatter_t2(const float* __restrict__ tmp2, const float* __restrict__ bias,
                             float* __restrict__ outp)
{
    long long i = (long long)blockIdx.x * 256 + threadIdx.x;
    long long tot = 128LL * 784 * 64;
    if (i >= tot) return;
    int n = (int)(i % 64); long long t2 = i / 64;
    int q = (int)(t2 % 784); int ib = (int)(t2 / 784);
    int d = n / 8, e = n % 8;
    int h = q / 28, w = q % 28;
    int t = ib / 8, b = ib % 8;
    outp[((long long)(b * 16 + t) * 224 + (h * 8 + d)) * 224 + (w * 8 + e)] = tmp2[i] + bias[0];
}

// ============================================================
// Host-side orchestration
// ============================================================
namespace {

struct Arena {
    char* base; size_t off;
    float* alloc(size_t nfloats) {
        size_t a = (off + 255) & ~(size_t)255;
        off = a + nfloats * sizeof(float);
        return (float*)(base + a);
    }
};

static inline dim3 eg(long long n) { return dim3((unsigned)((n + 255) / 256)); }

static void gemm(hipStream_t st, const float* A, const float* B, const float* bias,
                 const float* res, float* C, int M, int N, int K, int batch,
                 long long sA, long long sB, long long sC, int bTrans, float scale, int act)
{
    dim3 g((N + 63) / 64, (M + 63) / 64, batch);
    k_gemm<<<g, 256, 0, st>>>(A, B, bias, res, C, M, N, K, sA, sB, sC, bTrans, scale, act);
}

static void ln(hipStream_t st, const float* x, const float* g, const float* b,
               float* y, int rows, int D)
{
    k_layernorm<<<rows, 32, 0, st>>>(x, g, b, y, D);
}

static void mha(Arena& ar, hipStream_t st,
                const float* xq, const float* xkv, int B, int S, int L, int D, int nh,
                const float* wq, const float* bq, const float* wk, const float* bk,
                const float* wv, const float* bv, const float* wo, const float* bo,
                int causal, const float* res, float* out)
{
    const size_t mark = ar.off;
    const int hd = D / nh;
    float* q2 = ar.alloc((size_t)B * S * D);
    float* k2 = ar.alloc((size_t)B * L * D);
    float* v2 = ar.alloc((size_t)B * L * D);
    gemm(st, xq,  wq, bq, nullptr, q2, B * S, D, D, 1, 0, 0, 0, 0, 1.f, 0);
    gemm(st, xkv, wk, bk, nullptr, k2, B * L, D, D, 1, 0, 0, 0, 0, 1.f, 0);
    gemm(st, xkv, wv, bv, nullptr, v2, B * L, D, D, 1, 0, 0, 0, 0, 1.f, 0);

    float* qh = ar.alloc((size_t)B * S * D);
    float* kh = ar.alloc((size_t)B * L * D);
    float* vh = ar.alloc((size_t)B * L * D);
    k_to_heads<<<eg((long long)B * S * D), 256, 0, st>>>(q2, qh, B, S, nh, hd);
    k_to_heads<<<eg((long long)B * L * D), 256, 0, st>>>(k2, kh, B, L, nh, hd);
    k_to_heads<<<eg((long long)B * L * D), 256, 0, st>>>(v2, vh, B, L, nh, hd);

    float* sc = ar.alloc((size_t)B * nh * S * L);
    gemm(st, qh, kh, nullptr, nullptr, sc, S, L, hd, B * nh,
         (long long)S * hd, (long long)L * hd, (long long)S * L,
         0, 1.0f / sqrtf((float)hd), 0);
    k_softmax<<<B * nh * S, 32, 0, st>>>(sc, S, L, causal);

    float* oh = ar.alloc((size_t)B * nh * S * hd);
    gemm(st, sc, vh, nullptr, nullptr, oh, S, hd, L, B * nh,
         (long long)S * L, (long long)L * hd, (long long)S * hd, 1, 1.f, 0);

    float* c2 = ar.alloc((size_t)B * S * D);
    k_from_heads<<<eg((long long)B * S * D), 256, 0, st>>>(oh, c2, B, S, nh, hd);
    gemm(st, c2, wo, bo, res, out, B * S, D, D, 1, 0, 0, 0, 0, 1.f, 0);
    ar.off = mark;
}

// Input indices (setup_inputs dict order, nested dicts flattened in insertion order)
enum {
    IN_PIXEL = 0, IN_TEXT = 1, IN_TSI = 2,
    // vp
    VP_PATCH_W = 3, VP_CLS, VP_POS, VP_PRELN_G, VP_PRELN_B, VP_PROJ_W,
    VP_FC1_W, VP_FC1_B, VP_FC2_W, VP_FC2_B,
    VP_Q_W, VP_Q_B, VP_K_W, VP_K_B, VP_V_W, VP_V_B, VP_O_W, VP_O_B,
    VP_LN1_G, VP_LN1_B, VP_LN2_G, VP_LN2_B,
    // tp
    TP_TOK = 25, TP_POS, TP_FLN_G, TP_FLN_B, TP_PROJ_W,
    TP_FC1_W, TP_FC1_B, TP_FC2_W, TP_FC2_B,
    TP_Q_W, TP_Q_B, TP_K_W, TP_K_B, TP_V_W, TP_V_B, TP_O_W, TP_O_B,
    TP_LN1_G, TP_LN1_B, TP_LN2_G, TP_LN2_B,
    // dp
    DE_Q_W = 46, DE_Q_B, DE_K_W, DE_K_B, DE_V_W, DE_V_B, DE_O_W, DE_O_B,
    DE_LN1_G, DE_LN1_B, DE_LN2_G, DE_LN2_B,
    DE_FC1_W, DE_FC1_B, DE_FC2_W, DE_FC2_B,
    DS_Q_W, DS_Q_B, DS_K_W, DS_K_B, DS_V_W, DS_V_B, DS_O_W, DS_O_B,
    DC_Q_W, DC_Q_B, DC_K_W, DC_K_B, DC_V_W, DC_V_B, DC_O_W, DC_O_B,
    DD_LN1_G, DD_LN1_B, DD_LN2_G, DD_LN2_B, DD_LN3_G, DD_LN3_B,
    DD_FC1_W, DD_FC1_B, DD_FC2_W, DD_FC2_B,
    DP_EN_G, DP_EN_B, DP_DN_G, DP_DN_B,
    // fp
    FP_RED_W = 92, FP_RED_B, FP_FM_W, FP_FM_B, FP_FA_W, FP_FA_B,
    FP_C1_W, FP_C1_B, FP_T1_W, FP_T1_B, FP_T2_W, FP_T2_B,
    FP_LABEL_W, FP_LABEL_B,
    FB_FC1_W, FB_FC1_B, FB_FC2_W, FB_FC2_B,
    FB_Q_W, FB_Q_B, FB_K_W, FB_K_B, FB_V_W, FB_V_B, FB_O_W, FB_O_B,
    FB_LN1_G, FB_LN1_B, FB_LN2_G, FB_LN2_B
};

} // namespace

extern "C" void kernel_launch(void* const* d_in, const int* in_sizes, int n_in,
                              void* d_out, int out_size, void* d_ws, size_t ws_size,
                              hipStream_t stream)
{
    (void)in_sizes; (void)n_in; (void)out_size; (void)ws_size;
    hipStream_t st = stream;
    auto P = [&](int i) { return (const float*)d_in[i]; };

    Arena ar{(char*)d_ws, 0};

    // ---- persistent buffers ----
    float* patches = ar.alloc(392 * 3072);
    float* pe      = ar.alloc(392 * 768);
    float* xa      = ar.alloc(400 * 768);
    float* xb      = ar.alloc(400 * 768);
    float* hv      = ar.alloc(400 * 768);
    float* hs0     = ar.alloc(400 * 768);
    float* hs1     = ar.alloc(400 * 768);
    float* hs2     = ar.alloc(400 * 768);
    float* visfeat = ar.alloc(400 * 512);
    float* ea      = ar.alloc(400 * 512);
    float* ebuf    = ar.alloc(400 * 512);
    float* eT      = ar.alloc(400 * 512);
    float* memb    = ar.alloc(400 * 512);
    float* xt      = ar.alloc(128 * 512);
    float* xtb     = ar.alloc(128 * 512);
    float* ht      = ar.alloc(128 * 512);
    float* txtfeat = ar.alloc(128 * 512);
    float* da      = ar.alloc(128 * 512);
    float* dbuf    = ar.alloc(128 * 512);
    float* dT      = ar.alloc(128 * 512);
    float* lang    = ar.alloc(128 * 512);
    float* mulb    = ar.alloc(128 * 64);
    float* addb    = ar.alloc(128 * 64);
    float* ab0     = ar.alloc(400 * 64);
    float* ab1     = ar.alloc(400 * 64);
    float* ab2     = ar.alloc(400 * 64);
    float* afilm   = ar.alloc(16 * 50 * 8 * 64);
    float* aB      = ar.alloc(16 * 50 * 8 * 64);
    float* aT      = ar.alloc(16 * 50 * 8 * 64);
    float* hidbig  = ar.alloc(6400LL * 2048);          // shared FFN scratch (largest user)
    float* col     = ar.alloc(6272LL * 576);
    float* y1      = ar.alloc(6272LL * 64);
    float* tmp1    = ar.alloc(6272LL * 512);
    float* y2cl    = ar.alloc(100352LL * 32);
    float* tmp2    = ar.alloc(100352LL * 64);

    float* out_masks  = (float*)d_out;                 // [8,16,224,224]
    float* out_logits = (float*)d_out + 6422528LL;     // [8,16,49408]

    // ================= CLIP VISION =================
    k_patchify<<<eg(8LL * 49 * 3072), 256, 0, st>>>(P(IN_PIXEL), patches);
    gemm(st, patches, P(VP_PATCH_W), nullptr, nullptr, pe, 392, 768, 3072, 1, 0, 0, 0, 0, 1.f, 0);
    k_buildx<<<eg(8LL * 50 * 768), 256, 0, st>>>(pe, P(VP_CLS), P(VP_POS), xa);
    ln(st, xa, P(VP_PRELN_G), P(VP_PRELN_B), xa, 400, 768);

    for (int i = 0; i < 12; ++i) {
        const long long w = (long long)i * 768 * 768, bb = (long long)i * 768;
        const long long wf1 = (long long)i * 3072 * 768, bf1 = (long long)i * 3072;
        ln(st, xa, P(VP_LN1_G) + bb, P(VP_LN1_B) + bb, hv, 400, 768);
        mha(ar, st, hv, hv, 8, 50, 50, 768, 12,
            P(VP_Q_W) + w, P(VP_Q_B) + bb, P(VP_K_W) + w, P(VP_K_B) + bb,
            P(VP_V_W) + w, P(VP_V_B) + bb, P(VP_O_W) + w, P(VP_O_B) + bb,
            0, xa, xb);
        ln(st, xb, P(VP_LN2_G) + bb, P(VP_LN2_B) + bb, hv, 400, 768);
        gemm(st, hv, P(VP_FC1_W) + wf1, P(VP_FC1_B) + bf1, nullptr, hidbig, 400, 3072, 768, 1, 0, 0, 0, 0, 1.f, 1);
        gemm(st, hidbig, P(VP_FC2_W) + wf1, P(VP_FC2_B) + bb, xb, xa, 400, 768, 3072, 1, 0, 0, 0, 0, 1.f, 0);
        if (i == 2) k_copy<<<eg(400LL * 768), 256, 0, st>>>(xa, hs0, 400LL * 768);
        if (i == 5) k_copy<<<eg(400LL * 768), 256, 0, st>>>(xa, hs1, 400LL * 768);
        if (i == 8) k_copy<<<eg(400LL * 768), 256, 0, st>>>(xa, hs2, 400LL * 768);
    }
    gemm(st, xa, P(VP_PROJ_W), nullptr, nullptr, visfeat, 400, 512, 768, 1, 0, 0, 0, 0, 1.f, 0);

    // ================= CLIP TEXT =================
    k_embed<<<eg(8LL * 16 * 512), 256, 0, st>>>((const int*)d_in[IN_TEXT], P(TP_TOK), P(TP_POS), xt);
    for (int i = 0; i < 12; ++i) {
        const long long w = (long long)i * 512 * 512, bb = (long long)i * 512;
        const long long wf1 = (long long)i * 2048 * 512, bf1 = (long long)i * 2048;
        ln(st, xt, P(TP_LN1_G) + bb, P(TP_LN1_B) + bb, ht, 128, 512);
        mha(ar, st, ht, ht, 8, 16, 16, 512, 8,
            P(TP_Q_W) + w, P(TP_Q_B) + bb, P(TP_K_W) + w, P(TP_K_B) + bb,
            P(TP_V_W) + w, P(TP_V_B) + bb, P(TP_O_W) + w, P(TP_O_B) + bb,
            1, xt, xtb);
        ln(st, xtb, P(TP_LN2_G) + bb, P(TP_LN2_B) + bb, ht, 128, 512);
        gemm(st, ht, P(TP_FC1_W) + wf1, P(TP_FC1_B) + bf1, nullptr, hidbig, 128, 2048, 512, 1, 0, 0, 0, 0, 1.f, 1);
        gemm(st, hidbig, P(TP_FC2_W) + wf1, P(TP_FC2_B) + bb, xtb, xt, 128, 512, 2048, 1, 0, 0, 0, 0, 1.f, 0);
    }
    ln(st, xt, P(TP_FLN_G), P(TP_FLN_B), xt, 128, 512);
    gemm(st, xt, P(TP_PROJ_W), nullptr, nullptr, txtfeat, 128, 512, 512, 1, 0, 0, 0, 0, 1.f, 0);

    // ================= TORCH ENCODER (post-LN) =================
    k_copy<<<eg(400LL * 512), 256, 0, st>>>(visfeat, ea, 400LL * 512);
    for (int i = 0; i < 3; ++i) {
        const long long w = (long long)i * 512 * 512, bb = (long long)i * 512;
        const long long wf1 = (long long)i * 2048 * 512, bf1 = (long long)i * 2048;
        mha(ar, st, ea, ea, 8, 50, 50, 512, 4,
            P(DE_Q_W) + w, P(DE_Q_B) + bb, P(DE_K_W) + w, P(DE_K_B) + bb,
            P(DE_V_W) + w, P(DE_V_B) + bb, P(DE_O_W) + w, P(DE_O_B) + bb,
            0, ea, eT);
        ln(st, eT, P(DE_LN1_G) + bb, P(DE_LN1_B) + bb, ebuf, 400, 512);
        gemm(st, ebuf, P(DE_FC1_W) + wf1, P(DE_FC1_B) + bf1, nullptr, hidbig, 400, 2048, 512, 1, 0, 0, 0, 0, 1.f, 2);
        gemm(st, hidbig, P(DE_FC2_W) + wf1, P(DE_FC2_B) + bb, ebuf, eT, 400, 512, 2048, 1, 0, 0, 0, 0, 1.f, 0);
        ln(st, eT, P(DE_LN2_G) + bb, P(DE_LN2_B) + bb, ea, 400, 512);
    }
    ln(st, ea, P(DP_EN_G), P(DP_EN_B), memb, 400, 512);

    // ================= TORCH DECODER =================
    k_copy<<<eg(128LL * 512), 256, 0, st>>>(txtfeat, da, 128LL * 512);
    float* dx = da; float* dy = dbuf;
    for (int i = 0; i < 6; ++i) {
        const long long w = (long long)i * 512 * 512, bb = (long long)i * 512;
        const long long wf1 = (long long)i * 2048 * 512, bf1 = (long long)i * 2048;
        mha(ar, st, dx, dx, 8, 16, 16, 512, 4,
            P(DS_Q_W) + w, P(DS_Q_B) + bb, P(DS_K_W) + w, P(DS_K_B) + bb,
            P(DS_V_W) + w, P(DS_V_B) + bb, P(DS_O_W) + w, P(DS_O_B) + bb,
            1, dx, dT);
        ln(st, dT, P(DD_LN1_G) + bb, P(DD_LN1_B) + bb, dy, 128, 512);
        mha(ar, st, dy, memb, 8, 16, 50, 512, 4,
            P(DC_Q_W) + w, P(DC_Q_B) + bb, P(DC_K_W) + w, P(DC_K_B) + bb,
            P(DC_V_W) + w, P(DC_V_B) + bb, P(DC_O_W) + w, P(DC_O_B) + bb,
            0, dy, dT);
        ln(st, dT, P(DD_LN2_G) + bb, P(DD_LN2_B) + bb, dx, 128, 512);
        gemm(st, dx, P(DD_FC1_W) + wf1, P(DD_FC1_B) + bf1, nullptr, hidbig, 128, 2048, 512, 1, 0, 0, 0, 0, 1.f, 2);
        gemm(st, hidbig, P(DD_FC2_W) + wf1, P(DD_FC2_B) + bb, dx, dT, 128, 512, 2048, 1, 0, 0, 0, 0, 1.f, 0);
        ln(st, dT, P(DD_LN3_G) + bb, P(DD_LN3_B) + bb, dy, 128, 512);
        float* tmp = dx; dx = dy; dy = tmp;
    }
    ln(st, dx, P(DP_DN_G), P(DP_DN_B), lang, 128, 512);

    // ================= FiLM DECODE =================
    gemm(st, lang, P(FP_FM_W), P(FP_FM_B), nullptr, mulb, 128, 64, 512, 1, 0, 0, 0, 0, 1.f, 0);
    gemm(st, lang, P(FP_FA_W), P(FP_FA_B), nullptr, addb, 128, 64, 512, 1, 0, 0, 0, 0, 1.f, 0);
    float* hsarr[3] = {hs0, hs1, hs2};
    float* abarr[3] = {ab0, ab1, ab2};
    for (int i = 0; i < 3; ++i)
        gemm(st, hsarr[i], P(FP_RED_W) + (long long)i * 64 * 768, P(FP_RED_B) + i * 64,
             nullptr, abarr[i], 400, 64, 768, 1, 0, 0, 0, 0, 1.f, 0);

    k_zero<<<eg(16LL * 50 * 8 * 64), 256, 0, st>>>(afilm, 16LL * 50 * 8 * 64);
    for (int i = 0; i < 3; ++i) {
        const long long w = (long long)i * 64 * 64, bb = (long long)i * 64;
        const long long wf1 = (long long)i * 2048 * 64, bf1 = (long long)i * 2048;
        k_film<<<eg(16LL * 50 * 8 * 64), 256, 0, st>>>(afilm, abarr[i], mulb, addb);
        // post-LN block: attention over the batch axis (seq=8), batch=16*50
        mha(ar, st, afilm, afilm, 800, 8, 8, 64, 4,
            P(FB_Q_W) + w, P(FB_Q_B) + bb, P(FB_K_W) + w, P(FB_K_B) + bb,
            P(FB_V_W) + w, P(FB_V_B) + bb, P(FB_O_W) + w, P(FB_O_B) + bb,
            0, afilm, aT);
        ln(st, aT, P(FB_LN1_G) + bb, P(FB_LN1_B) + bb, aB, 6400, 64);
        gemm(st, aB, P(FB_FC1_W) + wf1, P(FB_FC1_B) + bf1, nullptr, hidbig, 6400, 2048, 64, 1, 0, 0, 0, 0, 1.f, 2);
        gemm(st, hidbig, P(FB_FC2_W) + wf1, P(FB_FC2_B) + bb, aB, aT, 6400, 64, 2048, 1, 0, 0, 0, 0, 1.f, 0);
        ln(st, aT, P(FB_LN2_G) + bb, P(FB_LN2_B) + bb, afilm, 6400, 64);
    }

    // ---- mask head ----
    k_im2col<<<eg(128LL * 49 * 576), 256, 0, st>>>(afilm, col);
    gemm(st, col, P(FP_C1_W), P(FP_C1_B), nullptr, y1, 6272, 64, 576, 1, 0, 0, 0, 0, 1.f, 2);
    gemm(st, y1, P(FP_T1_W), nullptr, nullptr, tmp1, 6272, 512, 64, 1, 0, 0, 0, 1, 1.f, 0);
    k_scatter_t1<<<eg(128LL * 784 * 32), 256, 0, st>>>(tmp1, P(FP_T1_B), y2cl);
    gemm(st, y2cl, P(FP_T2_W), nullptr, nullptr, tmp2, 100352, 64, 32, 1, 0, 0, 0, 1, 1.f, 0);
    k_scatter_t2<<<eg(128LL * 784 * 64), 256, 0, st>>>(tmp2, P(FP_T2_B), out_masks);

    // ---- logits ----
    gemm(st, lang, P(FP_LABEL_W), P(FP_LABEL_B), nullptr, out_logits, 128, 49408, 512, 1, 0, 0, 0, 0, 1.f, 0);
}
